// MultiHeadAttention_377957122345
// MI455X (gfx1250) — matrix operand order repaired and checked by reading, not compiled, not run on hardware
//
#include <hip/hip_runtime.h>
#include <hip/hip_bf16.h>
#include <cstdint>
#include <cstddef>

// ---------------------------------------------------------------------------
// Types for CDNA5 WMMA (wave32): bf16 16x16x32, f32 accumulate.
// ---------------------------------------------------------------------------
typedef __bf16 bf16;
typedef __attribute__((ext_vector_type(16))) __bf16 v16bf;
typedef __attribute__((ext_vector_type(8)))  __bf16 v8bf;
typedef __attribute__((ext_vector_type(4)))  __bf16 v4bf;
typedef __attribute__((ext_vector_type(2)))  __bf16 v2bf;
typedef __attribute__((ext_vector_type(8)))  float  v8f;

union Frag16 {
    v16bf v;
    struct { v8bf lo, hi; } h;
};

#define WMMA_BF16(A, B, C) \
    __builtin_amdgcn_wmma_f32_16x16x32_bf16(false, (A), false, (B), (short)0, (C), false, false)

// CDNA5 async global->LDS copy (ASYNCcnt-tracked, bypasses VGPRs).
#define ASYNC_LDS_B128(ldsaddr, gptr)                                     \
    asm volatile("global_load_async_to_lds_b128 %0, %1, off"              \
                 :: "v"(ldsaddr), "v"(gptr) : "memory")
#define WAIT_ASYNC0() asm volatile("s_wait_asynccnt 0" ::: "memory")

__device__ __forceinline__ unsigned lds_addr(const void* p) {
    return (unsigned)(size_t)p;   // LDS aperture: low 32 bits are the LDS offset
}

// Problem dims
constexpr int Bb = 16, Tt = 512, Cc = 1024, Hh = 16, Dd = 64;
constexpr int Mrows = Bb * Tt;   // 8192

// ---------------------------------------------------------------------------
// f32 -> bf16 conversion (vectorized, grid-stride)
// ---------------------------------------------------------------------------
__global__ __launch_bounds__(256) void cvt_bf16_kernel(const float* __restrict__ s,
                                                       bf16* __restrict__ d, int n4)
{
    int i = blockIdx.x * blockDim.x + threadIdx.x;
    int stride = gridDim.x * blockDim.x;
    for (; i < n4; i += stride) {
        float4 f = ((const float4*)s)[i];
        v4bf r = { (bf16)f.x, (bf16)f.y, (bf16)f.z, (bf16)f.w };
        ((v4bf*)d)[i] = r;
    }
}

// ---------------------------------------------------------------------------
// GEMM: Out[m,n] = sum_k A[m,k]*W[n,k] + bias[n]   (K = 1024 fixed)
// Block tile 128(M) x 128(N), 8 waves, each wave 32x64 (2x4 WMMA tiles).
// K-step 64 (2 WMMA chunks / LDS stage), double-buffered async-to-LDS fills.
// MODE 0: N=3072 fused QKV, scatter bf16 to (B,H,T,D), Q scaled by 1/8.
// MODE 1: N=1024, store f32 row-major.
// ---------------------------------------------------------------------------
template<int MODE>
__global__ __launch_bounds__(256) void gemm_bf16_kernel(
    const bf16* __restrict__ A,      // M x 1024, row-major (K contiguous)
    const bf16* __restrict__ W,      // N x 1024, row-major (K contiguous)
    const float* __restrict__ bias,  // N
    bf16* __restrict__ outQ, bf16* __restrict__ outK, bf16* __restrict__ outV,
    float* __restrict__ outF, int N)
{
    __shared__ bf16 As[2][128][64];   // 2 x 16 KB
    __shared__ bf16 Bs[2][128][64];   // 2 x 16 KB

    const int tid   = threadIdx.x;
    const int lane  = tid & 31;
    const int w     = tid >> 5;          // 0..7
    const int wm    = (w & 3) * 32;      // wave M offset in block tile
    const int wn    = (w >> 2) * 64;     // wave N offset in block tile
    const int mBase = blockIdx.x * 128;
    const int nBase = blockIdx.y * 128;

    const int lrow   = lane & 15;
    const int hi16   = lane >> 4;        // lane half
    const int khalfA = hi16 << 3;        // unused: A-frag also starts at khalfB (0 / 16)
    const int khalfB = hi16 << 4;        // B-frag k start within lane: 0 / 16

    v8f acc[2][4] = {};

    // cooperative fill mapping: each thread async-copies 32 elems of A and B
    const int frow  = tid >> 1;          // 0..127
    const int fhalf = (tid & 1) << 5;    // 0 / 32 elements

    const bf16* srcArow = A + (size_t)(mBase + frow) * 1024 + fhalf;
    const bf16* srcBrow = W + (size_t)(nBase + frow) * 1024 + fhalf;

    auto fill = [&](int buf, int k0) {
        const bf16* sa = srcArow + k0;
        unsigned la = lds_addr(&As[buf][frow][fhalf]);
        ASYNC_LDS_B128(la,       sa);
        ASYNC_LDS_B128(la + 16u, sa + 8);
        ASYNC_LDS_B128(la + 32u, sa + 16);
        ASYNC_LDS_B128(la + 48u, sa + 24);
        const bf16* sb = srcBrow + k0;
        unsigned lb = lds_addr(&Bs[buf][frow][fhalf]);
        ASYNC_LDS_B128(lb,       sb);
        ASYNC_LDS_B128(lb + 16u, sb + 8);
        ASYNC_LDS_B128(lb + 32u, sb + 16);
        ASYNC_LDS_B128(lb + 48u, sb + 24);
    };

    fill(0, 0);
    WAIT_ASYNC0();
    __syncthreads();

    int cur = 0;
    for (int k0 = 0; k0 < 1024; k0 += 64) {
        // prefetch next tile into the other buffer while we compute
        if (k0 + 64 < 1024) fill(cur ^ 1, k0 + 64);

        #pragma unroll
        for (int c = 0; c < 2; ++c) {          // two 32-wide k chunks
            Frag16 a[2], b[4];
            #pragma unroll
            for (int i = 0; i < 2; ++i) {
                const bf16* p = &As[cur][wm + i * 16 + lrow][c * 32];
                a[i].h.lo = *(const v8bf*)(p + khalfB);
                a[i].h.hi = *(const v8bf*)(p + khalfB + 8);
            }
            #pragma unroll
            for (int j = 0; j < 4; ++j) {
                const bf16* p = &Bs[cur][wn + j * 16 + lrow][c * 32];
                b[j].h.lo = *(const v8bf*)(p + khalfB);
                b[j].h.hi = *(const v8bf*)(p + khalfB + 8);
            }
            #pragma unroll
            for (int i = 0; i < 2; ++i)
                #pragma unroll
                for (int j = 0; j < 4; ++j)
                    acc[i][j] = WMMA_BF16(a[i].v, b[j].v, acc[i][j]);
        }

        WAIT_ASYNC0();        // next buffer's async fills landed
        __syncthreads();      // all waves done reading cur / filling next
        cur ^= 1;
    }

    // store: C-layout row = VGPR idx (+8 for upper lane half), col = lane&15
    const int rowoff = hi16 << 3;
    #pragma unroll
    for (int i = 0; i < 2; ++i) {
        #pragma unroll
        for (int j = 0; j < 4; ++j) {
            #pragma unroll
            for (int r = 0; r < 8; ++r) {
                int m = mBase + wm + i * 16 + r + rowoff;
                int n = nBase + wn + j * 16 + lrow;
                float val = acc[i][j][r] + bias[n];
                if (MODE == 0) {
                    int which = n >> 10;          // 0=Q 1=K 2=V
                    int nc = n & 1023;
                    int h = nc >> 6, d = nc & 63;
                    int bidx = m >> 9, t = m & 511;
                    size_t off = (((size_t)(bidx * Hh + h)) * Tt + t) * Dd + d;
                    if (which == 0)      outQ[off] = (bf16)(val * 0.125f); // fold 1/sqrt(64)
                    else if (which == 1) outK[off] = (bf16)val;
                    else                 outV[off] = (bf16)val;
                } else {
                    outF[(size_t)m * N + n] = val;
                }
            }
        }
    }
}

// ---------------------------------------------------------------------------
// Flash-style causal attention with relative-position bias.
// Grid: B*H*(T/64) blocks of 128 threads (4 waves x 16 q-rows).
// K tile and the 64x64 f32 bias tile are async-to-LDS; V transpose uses
// packed 2xbf16 ds_store_b32.
// ---------------------------------------------------------------------------
__global__ __launch_bounds__(128) void attn_kernel(
    const bf16* __restrict__ Q, const bf16* __restrict__ K,
    const bf16* __restrict__ V, const float* __restrict__ relb,
    bf16* __restrict__ O)
{
    __shared__ bf16  Ks[64][64];       // K tile, row-major [s][d]        8 KB
    __shared__ bf16  Vt[64][64];       // V tile, transposed [d][s]       8 KB
    __shared__ bf16  Ps[4][16][64];    // per-wave P relayout buffer      8 KB
    __shared__ float BiasS[64][64];    // rel-pos bias tile [t][s]       16 KB

    const int tid    = threadIdx.x;
    const int lane   = tid & 31;
    const int w      = tid >> 5;      // 0..3
    const int lrow   = lane & 15;
    const int hi16   = lane >> 4;
    const int khalfA = hi16 << 3;
    const int khalfB = hi16 << 4;
    const int rowoff = hi16 << 3;
    const int col    = lane & 15;

    const int bid  = blockIdx.x;
    const int qblk = bid & 7;
    const int h    = (bid >> 3) & 15;
    const int b    = bid >> 7;

    const size_t headoff = ((size_t)(b * Hh + h)) * Tt * Dd;
    const bf16* qh = Q + headoff;
    const bf16* kh = K + headoff;
    const bf16* vh = V + headoff;
    const float* biash = relb + (size_t)h * Tt * Tt;

    const int tb = qblk * 64 + w * 16;   // this wave's first q row

    // Q fragments held in registers for the whole kernel (scale pre-folded)
    Frag16 qa[2];
    {
        const bf16* qrow = qh + (size_t)(tb + lrow) * 64;
        qa[0].h.lo = *(const v8bf*)(qrow + khalfB);
        qa[0].h.hi = *(const v8bf*)(qrow + khalfB + 8);
        qa[1].h.lo = *(const v8bf*)(qrow + 32 + khalfB);
        qa[1].h.hi = *(const v8bf*)(qrow + 32 + khalfB + 8);
    }

    float mrow[8], lsum[8];
    #pragma unroll
    for (int r = 0; r < 8; ++r) { mrow[r] = -1e30f; lsum[r] = 0.0f; }
    v8f o[4] = {};

    const int fs   = tid >> 1;            // fill row 0..63
    const int fh   = (tid & 1) << 5;      // 0 / 32 elements
    const int vs0  = (tid & 31) << 1;     // V-transpose: s pair 0..62
    const int vdq  = (tid >> 5) << 4;     // V-transpose: d quarter 0/16/32/48

    for (int sb = 0; sb <= qblk * 64; sb += 64) {
        // ---- async fills: K tile (bf16) + bias tile (f32) ----
        {
            const bf16* ksrc = kh + (size_t)(sb + fs) * 64 + fh;
            unsigned lk = lds_addr(&Ks[fs][fh]);
            ASYNC_LDS_B128(lk,       ksrc);
            ASYNC_LDS_B128(lk + 16u, ksrc + 8);
            ASYNC_LDS_B128(lk + 32u, ksrc + 16);
            ASYNC_LDS_B128(lk + 48u, ksrc + 24);

            const float* bsrc = biash + (size_t)(qblk * 64 + fs) * Tt + sb + fh;
            unsigned lb = lds_addr(&BiasS[fs][fh]);
            #pragma unroll
            for (int q = 0; q < 8; ++q)
                ASYNC_LDS_B128(lb + (unsigned)(q * 16), bsrc + q * 4);

            // ---- V transpose: b128 reads, packed 2xbf16 b32 LDS stores ----
            const bf16* v0 = vh + (size_t)(sb + vs0) * 64 + vdq;
            v8bf r0a = *(const v8bf*)(v0);
            v8bf r0b = *(const v8bf*)(v0 + 8);
            v8bf r1a = *(const v8bf*)(v0 + 64);
            v8bf r1b = *(const v8bf*)(v0 + 64 + 8);
            #pragma unroll
            for (int d = 0; d < 8; ++d) {
                v2bf p0; p0[0] = r0a[d]; p0[1] = r1a[d];
                *(v2bf*)&Vt[vdq + d][vs0] = p0;
                v2bf p1; p1[0] = r0b[d]; p1[1] = r1b[d];
                *(v2bf*)&Vt[vdq + 8 + d][vs0] = p1;
            }
        }
        WAIT_ASYNC0();
        __syncthreads();

        // ---- S = Q K^T  (16 x 64 per wave) ----
        v8f s4[4] = {};
        #pragma unroll
        for (int j = 0; j < 4; ++j) {
            const bf16* krow = &Ks[j * 16 + lrow][0];
            Frag16 kb0, kb1;
            kb0.h.lo = *(const v8bf*)(krow + khalfB);
            kb0.h.hi = *(const v8bf*)(krow + khalfB + 8);
            kb1.h.lo = *(const v8bf*)(krow + 32 + khalfB);
            kb1.h.hi = *(const v8bf*)(krow + 32 + khalfB + 8);
            s4[j] = WMMA_BF16(qa[0].v, kb0.v, s4[j]);
            s4[j] = WMMA_BF16(qa[1].v, kb1.v, s4[j]);
        }

        // ---- rel-pos bias (from LDS) + causal mask ----
        #pragma unroll
        for (int j = 0; j < 4; ++j) {
            int scol = sb + j * 16 + col;
            #pragma unroll
            for (int r = 0; r < 8; ++r) {
                int t = tb + r + rowoff;
                float vv = s4[j][r] + BiasS[w * 16 + r + rowoff][j * 16 + col];
                s4[j][r] = (scol <= t) ? vv : -1e30f;
            }
        }

        // ---- online softmax (rows live across lanes of one 16-lane half) ----
        #pragma unroll
        for (int r = 0; r < 8; ++r) {
            float rmax = fmaxf(fmaxf(s4[0][r], s4[1][r]), fmaxf(s4[2][r], s4[3][r]));
            rmax = fmaxf(rmax, __shfl_xor(rmax, 1, 32));
            rmax = fmaxf(rmax, __shfl_xor(rmax, 2, 32));
            rmax = fmaxf(rmax, __shfl_xor(rmax, 4, 32));
            rmax = fmaxf(rmax, __shfl_xor(rmax, 8, 32));
            float mnew = fmaxf(mrow[r], rmax);
            float p0 = __expf(s4[0][r] - mnew);
            float p1 = __expf(s4[1][r] - mnew);
            float p2 = __expf(s4[2][r] - mnew);
            float p3 = __expf(s4[3][r] - mnew);
            float rsum = p0 + p1 + p2 + p3;
            rsum += __shfl_xor(rsum, 1, 32);
            rsum += __shfl_xor(rsum, 2, 32);
            rsum += __shfl_xor(rsum, 4, 32);
            rsum += __shfl_xor(rsum, 8, 32);
            float corr = __expf(mrow[r] - mnew);
            lsum[r] = lsum[r] * corr + rsum;
            mrow[r] = mnew;
            #pragma unroll
            for (int j = 0; j < 4; ++j) o[j][r] *= corr;
            int trl = r + rowoff;
            Ps[w][trl][0 * 16 + col] = (bf16)p0;
            Ps[w][trl][1 * 16 + col] = (bf16)p1;
            Ps[w][trl][2 * 16 + col] = (bf16)p2;
            Ps[w][trl][3 * 16 + col] = (bf16)p3;
        }

        // ---- O += P @ V ----
        {
            Frag16 pa0, pa1;
            const bf16* prow = &Ps[w][lrow][0];
            pa0.h.lo = *(const v8bf*)(prow + khalfB);
            pa0.h.hi = *(const v8bf*)(prow + khalfB + 8);
            pa1.h.lo = *(const v8bf*)(prow + 32 + khalfB);
            pa1.h.hi = *(const v8bf*)(prow + 32 + khalfB + 8);
            #pragma unroll
            for (int j = 0; j < 4; ++j) {
                const bf16* vrow = &Vt[j * 16 + lrow][0];
                Frag16 vb0, vb1;
                vb0.h.lo = *(const v8bf*)(vrow + khalfB);
                vb0.h.hi = *(const v8bf*)(vrow + khalfB + 8);
                vb1.h.lo = *(const v8bf*)(vrow + 32 + khalfB);
                vb1.h.hi = *(const v8bf*)(vrow + 32 + khalfB + 8);
                o[j] = WMMA_BF16(pa0.v, vb0.v, o[j]);
                o[j] = WMMA_BF16(pa1.v, vb1.v, o[j]);
            }
        }
        __syncthreads();
    }

    // ---- normalize + store to (B, T, H*D) bf16 (A operand of final GEMM) ----
    #pragma unroll
    for (int j = 0; j < 4; ++j) {
        #pragma unroll
        for (int r = 0; r < 8; ++r) {
            int t = tb + r + rowoff;
            int d = j * 16 + col;
            float val = o[j][r] / lsum[r];
            O[((size_t)(b * Tt + t)) * Cc + h * Dd + d] = (bf16)val;
        }
    }
}

// ---------------------------------------------------------------------------
// Launcher
// ---------------------------------------------------------------------------
extern "C" void kernel_launch(void* const* d_in, const int* in_sizes, int n_in,
                              void* d_out, int out_size, void* d_ws, size_t ws_size,
                              hipStream_t stream)
{
    (void)in_sizes; (void)n_in; (void)out_size; (void)ws_size;
    const float* x    = (const float*)d_in[0];
    const float* Wq   = (const float*)d_in[1];
    const float* bq   = (const float*)d_in[2];
    const float* Wk   = (const float*)d_in[3];
    const float* bk   = (const float*)d_in[4];
    const float* Wv   = (const float*)d_in[5];
    const float* bv   = (const float*)d_in[6];
    const float* Wo   = (const float*)d_in[7];
    const float* bo   = (const float*)d_in[8];
    const float* relb = (const float*)d_in[9];
    float* out = (float*)d_out;

    // workspace layout (~89 MB total)
    const size_t MB = 1024ull * 1024ull;
    char* ws = (char*)d_ws;
    bf16*  Xb      = (bf16*)(ws + 0 * MB);   // 16 MB  x in bf16
    bf16*  Wqkvb   = (bf16*)(ws + 16 * MB);  //  6 MB  [Wq;Wk;Wv] bf16
    bf16*  Wob     = (bf16*)(ws + 22 * MB);  //  2 MB  Wo bf16
    float* biasqkv = (float*)(ws + 24 * MB); // 12 KB  [bq;bk;bv]
    bf16*  Qb      = (bf16*)(ws + 25 * MB);  // 16 MB  (B,H,T,D)
    bf16*  Kb      = (bf16*)(ws + 41 * MB);  // 16 MB
    bf16*  Vb      = (bf16*)(ws + 57 * MB);  // 16 MB
    bf16*  Ab      = (bf16*)(ws + 73 * MB);  // 16 MB  attn out (B,T,C)

    // f32 -> bf16 (vectorized by 4)
    cvt_bf16_kernel<<<1024, 256, 0, stream>>>(x,  Xb,               (Mrows * Cc) / 4);
    cvt_bf16_kernel<<<256,  256, 0, stream>>>(Wq, Wqkvb,            (Cc * Cc) / 4);
    cvt_bf16_kernel<<<256,  256, 0, stream>>>(Wk, Wqkvb + Cc * Cc,  (Cc * Cc) / 4);
    cvt_bf16_kernel<<<256,  256, 0, stream>>>(Wv, Wqkvb + 2*Cc*Cc,  (Cc * Cc) / 4);
    cvt_bf16_kernel<<<256,  256, 0, stream>>>(Wo, Wob,              (Cc * Cc) / 4);

    hipMemcpyAsync(biasqkv,          bq, Cc * sizeof(float), hipMemcpyDeviceToDevice, stream);
    hipMemcpyAsync(biasqkv + Cc,     bk, Cc * sizeof(float), hipMemcpyDeviceToDevice, stream);
    hipMemcpyAsync(biasqkv + 2 * Cc, bv, Cc * sizeof(float), hipMemcpyDeviceToDevice, stream);

    // fused QKV projection: M=8192, N=3072
    gemm_bf16_kernel<0><<<dim3(Mrows / 128, (3 * Cc) / 128), 256, 0, stream>>>(
        Xb, Wqkvb, biasqkv, Qb, Kb, Vb, nullptr, 3 * Cc);

    // attention: 16*16*8 = 2048 blocks
    attn_kernel<<<Bb * Hh * (Tt / 64), 128, 0, stream>>>(Qb, Kb, Vb, relb, Ab);

    // output projection: M=8192, N=1024, f32 out
    gemm_bf16_kernel<1><<<dim3(Mrows / 128, Cc / 128), 256, 0, stream>>>(
        Ab, Wob, bo, nullptr, nullptr, nullptr, out, Cc);
}
